// SimilarityModel_49503793053849
// MI455X (gfx1250) — compile-verified, matrix operand order checked by reading, not run on hardware
//
#include <hip/hip_runtime.h>

// ---------------------------------------------------------------------------
// MI455X (gfx1250) implementation.
// Algebraic reduction: the 1x1-conv head means out = attn@V is never needed:
//   seg_i = sum_j attn_ij * w_j + conv_b,  w_j = O_j . u + cb,
//   u = Wv^T conv_w, cb = bv . conv_w.
// Dominant compute: S = Q K^T (322 GFLOP bf16 WMMA) + Q/K projections
// (52 GFLOP). Split-K flash attention (7 key splits) for grid parallelism;
// per-split (m, l, sum e*w) partials merged with log-sum-exp.
// Attention staging uses gfx1250 async global->LDS loads (ASYNCcnt) with
// double-buffered LDS tiles; falls back to plain copies if builtins absent.
// ---------------------------------------------------------------------------

typedef __bf16 bf16;
typedef __attribute__((ext_vector_type(16))) __bf16        v16bf;
typedef __attribute__((ext_vector_type(8)))  float         v8f;
typedef __attribute__((ext_vector_type(4)))  unsigned int  u32x4;
typedef __attribute__((__vector_size__(16))) int           i32x4v;

#define DCH    1024          // mapped_dim
#define NTOK   12544         // 112*112 tokens
#define KPAD   40            // padded LDS row stride in halfs (bank spread)
#define NKT    (NTOK / 128)  // 98 key tiles
#define KSPLIT 7             // key-dim splits (98 = 7 * 14)
#define KT_PER (NKT / KSPLIT)

#if defined(__gfx1250__) &&                                        \
    __has_builtin(__builtin_amdgcn_global_load_async_to_lds_b128) && \
    __has_builtin(__builtin_amdgcn_s_wait_asynccnt)
#define HAVE_ASYNC_LDS 1
#else
#define HAVE_ASYNC_LDS 0
#endif

// The async-load builtin expects v4i pointers in addrspace(1)/(3).
#define AS1P(p) ((__attribute__((address_space(1))) i32x4v*)(p))
#define AS3P(p) ((__attribute__((address_space(3))) i32x4v*)(p))

// Load one 16x32 bf16 WMMA fragment row (A-matrix layout, ISA 7.12.2):
// lane L: row = L&15, khalf g = L>>4; VGPR v packs K = (v>>2)*16 + g*8 + (v&3)*2.
// => elements 0..7 = halfs [g*8, g*8+8), elements 8..15 = halfs [16+g*8, +8).
// B (32x16) uses the mirrored packing, reading identically from a row-major
// LDS tile whose rows are the 16 B-columns.
__device__ __forceinline__ v16bf load_frag_row(const bf16* rowp, int g) {
  union { u32x4 u[2]; v16bf v; } f;
  f.u[0] = *(const u32x4*)(rowp + g * 8);
  f.u[1] = *(const u32x4*)(rowp + 16 + g * 8);
  return f.v;
}

__device__ __forceinline__ v8f wmma_bf16(v16bf a, v16bf b, v8f c) {
  return __builtin_amdgcn_wmma_f32_16x16x32_bf16(false, a, false, b, (short)0, c,
                                                 false, false);
}

// ---------------- kernel 1: u = Wv^T conv_w ; cb = bv . conv_w ----------------
__global__ __launch_bounds__(256) void prep_u_kernel(
    const float* __restrict__ Wv, const float* __restrict__ cw,
    const float* __restrict__ bv, float* __restrict__ u, float* __restrict__ cb) {
  int d = blockIdx.x * 256 + threadIdx.x;
  if (d < DCH) {
    float s = 0.f;
    for (int e = 0; e < DCH; ++e) s += Wv[(size_t)e * DCH + d] * cw[e];
    u[d] = s;
  }
  if (d == 0) {
    float c = 0.f;
    for (int e = 0; e < DCH; ++e) c += bv[e] * cw[e];
    cb[0] = c;
  }
}

// ---------------- kernel 2: Wq/Wk f32 -> bf16 ----------------
__global__ __launch_bounds__(256) void cvt_w_kernel(
    const float* __restrict__ Wq, const float* __restrict__ Wk,
    bf16* __restrict__ Wqb, bf16* __restrict__ Wkb) {
  size_t i = (size_t)blockIdx.x * 256 + threadIdx.x;  // grid covers D*D exactly
  Wqb[i] = (bf16)Wq[i];
  Wkb[i] = (bf16)Wk[i];
}

// ---------------- kernel 3: transpose (D,N)->(N,D), add patch_embed gather ----
__global__ __launch_bounds__(256) void pack_kernel(
    const float* __restrict__ feat, const int* __restrict__ sig,
    const float* __restrict__ pe, bf16* __restrict__ Abf, bf16* __restrict__ Obf) {
  __shared__ float tile[32][33];
  int tx = threadIdx.x, ty = threadIdx.y;  // (32, 8)
  int n0 = blockIdx.x * 32, d0 = blockIdx.y * 32;
#pragma unroll
  for (int i = 0; i < 4; ++i)
    tile[ty + i * 8][tx] = feat[(size_t)(d0 + ty + i * 8) * NTOK + n0 + tx];
  __syncthreads();
#pragma unroll
  for (int i = 0; i < 4; ++i) {
    int n = n0 + ty + i * 8;
    int d = d0 + tx;
    float v = tile[tx][ty + i * 8];
    int s = sig[n];
    Obf[(size_t)n * DCH + d] = (bf16)v;
    Abf[(size_t)n * DCH + d] = (bf16)(v + pe[(size_t)s * DCH + d]);
  }
}

// ---------------- kernel 4: w[n] = O_n . u + cb ----------------
__global__ __launch_bounds__(256) void wvec_kernel(
    const float* __restrict__ feat, const float* __restrict__ u,
    const float* __restrict__ cb, float* __restrict__ wv) {
  int n = blockIdx.x * 256 + threadIdx.x;  // grid covers NTOK exactly
  float s = 0.f;
  for (int d = 0; d < DCH; ++d) s += feat[(size_t)d * NTOK + n] * u[d];
  wv[n] = s + cb[0];
}

// ---------------- kernel 5: Q = A Wq^T + bq ; K = O Wk^T + bk (bf16 out) ------
// 128x128 block tile, BK=32, 8 waves in 4x2 grid, 2x4 WMMA tiles per wave.
__global__ __launch_bounds__(256) void qk_gemm_kernel(
    const bf16* __restrict__ Abf, const bf16* __restrict__ Obf,
    const bf16* __restrict__ Wqb, const bf16* __restrict__ Wkb,
    const float* __restrict__ bq, const float* __restrict__ bk,
    bf16* __restrict__ Qbf, bf16* __restrict__ Kbf) {
  const bf16* X;
  const bf16* W;
  const float* bias;
  bf16* Out;
  if (blockIdx.z == 0) { X = Abf; W = Wqb; bias = bq; Out = Qbf; }
  else                 { X = Obf; W = Wkb; bias = bk; Out = Kbf; }

  __shared__ bf16 Xs[128 * KPAD];
  __shared__ bf16 Ws[128 * KPAD];

  int tid = threadIdx.x;
  int wave = tid >> 5, lane = tid & 31;
  int g = lane >> 4, ln = lane & 15;
  int wm = wave >> 1, wn = wave & 1;  // 4x2 wave grid
  int m0 = blockIdx.x * 128;          // token rows
  int n0 = blockIdx.y * 128;          // output-channel cols

  v8f C[2][4];
  v8f vzero = {0.f, 0.f, 0.f, 0.f, 0.f, 0.f, 0.f, 0.f};
#pragma unroll
  for (int s = 0; s < 2; ++s)
#pragma unroll
    for (int t = 0; t < 4; ++t) C[s][t] = vzero;

  for (int k0 = 0; k0 < DCH; k0 += 32) {
    __syncthreads();
#pragma unroll
    for (int i = 0; i < 2; ++i) {
      int lin = tid + i * 256;  // 0..511 -> 128 rows x 4 chunks
      int row = lin >> 2, ch = lin & 3;
      *(u32x4*)(Xs + row * KPAD + ch * 8) =
          *(const u32x4*)(X + (size_t)(m0 + row) * DCH + k0 + ch * 8);
      *(u32x4*)(Ws + row * KPAD + ch * 8) =
          *(const u32x4*)(W + (size_t)(n0 + row) * DCH + k0 + ch * 8);
    }
    __syncthreads();
    v16bf a0 = load_frag_row(Xs + (wm * 32 + ln) * KPAD, g);
    v16bf a1 = load_frag_row(Xs + (wm * 32 + 16 + ln) * KPAD, g);
#pragma unroll
    for (int t = 0; t < 4; ++t) {
      v16bf b = load_frag_row(Ws + (wn * 64 + t * 16 + ln) * KPAD, g);
      C[0][t] = wmma_bf16(a0, b, C[0][t]);
      C[1][t] = wmma_bf16(a1, b, C[1][t]);
    }
  }

#pragma unroll
  for (int s = 0; s < 2; ++s)
#pragma unroll
    for (int t = 0; t < 4; ++t) {
      int col = n0 + wn * 64 + t * 16 + ln;
      float bval = bias[col];
#pragma unroll
      for (int r = 0; r < 8; ++r) {
        int row = m0 + wm * 32 + s * 16 + g * 8 + r;
        Out[(size_t)row * DCH + col] = (bf16)(C[s][t][r] + bval);
      }
    }
}

// ---------------- kernel 6: split-K online-softmax attention ------------------
// Grid (98 qblocks, 7 ksplits), 128 threads (4 waves). Wave owns 32 query
// rows: 2 A-frags x 8 B-frags = 16 WMMAs per 32-deep k-step. Double-buffered
// LDS tiles filled with async global->LDS loads (ASYNCcnt pipelined).
__global__ __launch_bounds__(128) void attn_kernel(
    const bf16* __restrict__ Qbf, const bf16* __restrict__ Kbf,
    const float* __restrict__ wvec, float* __restrict__ Pm,
    float* __restrict__ Pl, float* __restrict__ Pa) {
  __shared__ bf16 Qs[2][128 * KPAD];
  __shared__ bf16 Ks[2][128 * KPAD];

  int tid = threadIdx.x;
  int wave = tid >> 5, lane = tid & 31;
  int g = lane >> 4, ln = lane & 15;
  int qblk = blockIdx.x, ks = blockIdx.y;
  int kt0 = ks * KT_PER;

  const size_t qbase = (size_t)(qblk * 128) * DCH;
  const float scale = 0.03125f;  // 1/sqrt(1024)
  v8f vzero = {0.f, 0.f, 0.f, 0.f, 0.f, 0.f, 0.f, 0.f};

  float m_run[2][8], l_run[2][8], a_run[2][8];
  v8f S[2][8];
#pragma unroll
  for (int s = 0; s < 2; ++s)
#pragma unroll
    for (int r = 0; r < 8; ++r) {
      m_run[s][r] = -1e30f;
      l_run[s][r] = 0.f;
      a_run[s][r] = 0.f;
    }
#pragma unroll
  for (int s = 0; s < 2; ++s)
#pragma unroll
    for (int t = 0; t < 8; ++t) S[s][t] = vzero;

  // Issue one 32-deep k-slice of Q and K tiles into LDS buffer `buf`.
  auto issue = [&](int step, int buf) {
    int kt = kt0 + (step >> 5);
    int k0 = (step & 31) * 32;
    size_t kbase = (size_t)(kt * 128) * DCH;
#pragma unroll
    for (int i = 0; i < 4; ++i) {
      int lin = tid + i * 128;  // 0..511 -> 128 rows x 4 chunks
      int row = lin >> 2, ch = lin & 3;
      const bf16* gq = Qbf + qbase + (size_t)row * DCH + k0 + ch * 8;
      const bf16* gk = Kbf + kbase + (size_t)row * DCH + k0 + ch * 8;
      bf16* lq = &Qs[buf][row * KPAD + ch * 8];
      bf16* lk = &Ks[buf][row * KPAD + ch * 8];
#if HAVE_ASYNC_LDS
      __builtin_amdgcn_global_load_async_to_lds_b128(AS1P(gq), AS3P(lq), 0, 0);
      __builtin_amdgcn_global_load_async_to_lds_b128(AS1P(gk), AS3P(lk), 0, 0);
#else
      *(u32x4*)lq = *(const u32x4*)gq;
      *(u32x4*)lk = *(const u32x4*)gk;
#endif
    }
  };

  const int TOTAL = KT_PER * 32;  // 14 key tiles x 32 k-steps
  issue(0, 0);
  for (int step = 0; step < TOTAL; ++step) {
    int buf = step & 1;
    if (step + 1 < TOTAL) issue(step + 1, buf ^ 1);
#if HAVE_ASYNC_LDS
    if (step + 1 < TOTAL) __builtin_amdgcn_s_wait_asynccnt(8);
    else                  __builtin_amdgcn_s_wait_asynccnt(0);
#endif
    __syncthreads();

    v16bf a0 = load_frag_row(&Qs[buf][(wave * 32 + ln) * KPAD], g);
    v16bf a1 = load_frag_row(&Qs[buf][(wave * 32 + 16 + ln) * KPAD], g);
#pragma unroll
    for (int t = 0; t < 8; ++t) {
      v16bf b = load_frag_row(&Ks[buf][(t * 16 + ln) * KPAD], g);
      S[0][t] = wmma_bf16(a0, b, S[0][t]);
      S[1][t] = wmma_bf16(a1, b, S[1][t]);
    }

    if ((step & 31) == 31) {  // finished one 128-key tile: online softmax
      int kt = kt0 + (step >> 5);
#pragma unroll
      for (int s = 0; s < 2; ++s) {
        float pm[8];
#pragma unroll
        for (int r = 0; r < 8; ++r) pm[r] = -1e30f;
#pragma unroll
        for (int t = 0; t < 8; ++t)
#pragma unroll
          for (int r = 0; r < 8; ++r) {
            S[s][t][r] *= scale;
            pm[r] = fmaxf(pm[r], S[s][t][r]);
          }
#pragma unroll
        for (int msk = 1; msk < 16; msk <<= 1)
#pragma unroll
          for (int r = 0; r < 8; ++r)
            pm[r] = fmaxf(pm[r], __shfl_xor(pm[r], msk, 32));

        float mn[8];
#pragma unroll
        for (int r = 0; r < 8; ++r) {
          mn[r] = fmaxf(m_run[s][r], pm[r]);
          float sc = __expf(m_run[s][r] - mn[r]);
          l_run[s][r] *= sc;
          a_run[s][r] *= sc;
          m_run[s][r] = mn[r];
        }
        float se[8], sw[8];
#pragma unroll
        for (int r = 0; r < 8; ++r) { se[r] = 0.f; sw[r] = 0.f; }
#pragma unroll
        for (int t = 0; t < 8; ++t) {
          float wj = wvec[kt * 128 + t * 16 + ln];
#pragma unroll
          for (int r = 0; r < 8; ++r) {
            float e = __expf(S[s][t][r] - mn[r]);
            se[r] += e;
            sw[r] += e * wj;
          }
        }
#pragma unroll
        for (int msk = 1; msk < 16; msk <<= 1)
#pragma unroll
          for (int r = 0; r < 8; ++r) {
            se[r] += __shfl_xor(se[r], msk, 32);
            sw[r] += __shfl_xor(sw[r], msk, 32);
          }
#pragma unroll
        for (int r = 0; r < 8; ++r) {
          l_run[s][r] += se[r];
          a_run[s][r] += sw[r];
        }
#pragma unroll
        for (int t = 0; t < 8; ++t) S[s][t] = vzero;
      }
    }
    __syncthreads();
  }

  if (ln == 0) {
#pragma unroll
    for (int s = 0; s < 2; ++s)
#pragma unroll
      for (int r = 0; r < 8; ++r) {
        int row = qblk * 128 + wave * 32 + s * 16 + g * 8 + r;
        size_t idx = (size_t)ks * NTOK + row;
        Pm[idx] = m_run[s][r];
        Pl[idx] = l_run[s][r];
        Pa[idx] = a_run[s][r];
      }
  }
}

// ---------------- kernel 7: merge split-K partials, sigmoid epilogue ----------
__global__ __launch_bounds__(256) void merge_kernel(
    const float* __restrict__ Pm, const float* __restrict__ Pl,
    const float* __restrict__ Pa, const float* __restrict__ convb,
    float* __restrict__ out) {
  int n = blockIdx.x * 256 + threadIdx.x;  // grid covers NTOK exactly
  float m = -1e30f;
#pragma unroll
  for (int k = 0; k < KSPLIT; ++k) m = fmaxf(m, Pm[(size_t)k * NTOK + n]);
  float l = 0.f, a = 0.f;
#pragma unroll
  for (int k = 0; k < KSPLIT; ++k) {
    float e = __expf(Pm[(size_t)k * NTOK + n] - m);
    l += Pl[(size_t)k * NTOK + n] * e;
    a += Pa[(size_t)k * NTOK + n] * e;
  }
  float v = a / l + convb[0];
  out[n] = 1.f / (1.f + __expf(-v));
}

// ---------------------------------------------------------------------------
extern "C" void kernel_launch(void* const* d_in, const int* in_sizes, int n_in,
                              void* d_out, int out_size, void* d_ws, size_t ws_size,
                              hipStream_t stream) {
  const int*   sig  = (const int*)d_in[0];
  const float* feat = (const float*)d_in[1];
  // d_in[2] = roi_mask: unused by the reference math
  const float* pe   = (const float*)d_in[3];
  const float* Wq   = (const float*)d_in[4];
  const float* bq   = (const float*)d_in[5];
  const float* Wk   = (const float*)d_in[6];
  const float* bk   = (const float*)d_in[7];
  const float* Wv   = (const float*)d_in[8];
  const float* bv   = (const float*)d_in[9];
  const float* cw   = (const float*)d_in[10];
  const float* cbv  = (const float*)d_in[11];
  float* out = (float*)d_out;

  char* ws = (char*)d_ws;
  size_t off = 0;
  auto carve = [&](size_t bytes) -> void* {
    void* p = ws + off;
    off = (off + bytes + 255) & ~(size_t)255;
    return p;
  };
  bf16* Abf  = (bf16*)carve((size_t)NTOK * DCH * sizeof(bf16));
  bf16* Obf  = (bf16*)carve((size_t)NTOK * DCH * sizeof(bf16));
  bf16* Qbf  = (bf16*)carve((size_t)NTOK * DCH * sizeof(bf16));
  bf16* Kbf  = (bf16*)carve((size_t)NTOK * DCH * sizeof(bf16));
  bf16* Wqb  = (bf16*)carve((size_t)DCH * DCH * sizeof(bf16));
  bf16* Wkb  = (bf16*)carve((size_t)DCH * DCH * sizeof(bf16));
  float* u   = (float*)carve(DCH * sizeof(float));
  float* wv  = (float*)carve(NTOK * sizeof(float));
  float* cb  = (float*)carve(sizeof(float));
  float* Pm  = (float*)carve((size_t)KSPLIT * NTOK * sizeof(float));
  float* Pl  = (float*)carve((size_t)KSPLIT * NTOK * sizeof(float));
  float* Pa  = (float*)carve((size_t)KSPLIT * NTOK * sizeof(float));

  prep_u_kernel<<<dim3(DCH / 256), dim3(256), 0, stream>>>(Wv, cw, bv, u, cb);
  cvt_w_kernel<<<dim3((DCH * DCH) / 256), dim3(256), 0, stream>>>(Wq, Wk, Wqb, Wkb);
  pack_kernel<<<dim3(NTOK / 32, DCH / 32), dim3(32, 8), 0, stream>>>(feat, sig, pe,
                                                                     Abf, Obf);
  wvec_kernel<<<dim3(NTOK / 256), dim3(256), 0, stream>>>(feat, u, cb, wv);
  qk_gemm_kernel<<<dim3(NTOK / 128, DCH / 128, 2), dim3(256), 0, stream>>>(
      Abf, Obf, Wqb, Wkb, bq, bk, Qbf, Kbf);
  attn_kernel<<<dim3(NTOK / 128, KSPLIT), dim3(128), 0, stream>>>(Qbf, Kbf, wv,
                                                                  Pm, Pl, Pa);
  merge_kernel<<<dim3(NTOK / 256), dim3(256), 0, stream>>>(Pm, Pl, Pa, cbv, out);
}
